// Net_87445534147142
// MI455X (gfx1250) — matrix operand order repaired and checked by reading, compiled, not run on hardware
//
#include <hip/hip_runtime.h>
#include <math.h>

typedef __attribute__((ext_vector_type(16))) _Float16 v16h;
typedef __attribute__((ext_vector_type(2)))  __fp16   v2fp16;
typedef __attribute__((ext_vector_type(8)))  float    v8f;

#define LDA 34  // 32 K + 2 pad halves; 17-bank row stride -> conflict-free gathers

// ---------------- utility kernels ----------------

__global__ void fill4_kernel(float4* __restrict__ p, float v, long long count4) {
    long long i = (long long)blockIdx.x * blockDim.x + threadIdx.x;
    if (i < count4) { float4 q = {v, v, v, v}; p[i] = q; }
}

// ---------------- edge scatter: basis-weighted source features -> acc buckets ----------------
// acc layout: acc[node][k*in_c + c]  (so the einsum becomes a single dense GEMM)

__global__ void spline_scatter(const float* __restrict__ x,
                               const int* __restrict__ src,
                               const int* __restrict__ dst,
                               const float* __restrict__ pseudo,
                               float* __restrict__ acc,
                               float* __restrict__ deg,
                               int E, int in_c, int cshift) {
    int t = blockIdx.x * blockDim.x + threadIdx.x;
    if (t >= (E << cshift)) return;
    int e = t >> cshift;
    int c = t & (in_c - 1);

    float v0 = pseudo[e * 3 + 0] * 4.0f;
    float v1 = pseudo[e * 3 + 1] * 4.0f;
    float v2 = pseudo[e * 3 + 2] * 4.0f;
    float b0 = fminf(floorf(v0), 3.0f);
    float b1 = fminf(floorf(v1), 3.0f);
    float b2 = fminf(floorf(v2), 3.0f);
    float f0 = v0 - b0, f1 = v1 - b1, f2 = v2 - b2;
    int i0 = (int)b0, i1 = (int)b1, i2 = (int)b2;

    int d = dst[e];
    float xv = x[((long long)src[e] << cshift) + c];
    if (c == 0) atomicAdd(&deg[d], 1.0f);

    long long base = (long long)d * 125;
#pragma unroll
    for (int s = 0; s < 8; ++s) {
        float w0 = (s & 1) ? f0 : 1.0f - f0;
        float w1 = (s & 2) ? f1 : 1.0f - f1;
        float w2 = (s & 4) ? f2 : 1.0f - f2;
        int widx = (i0 + (s & 1)) + 5 * (i1 + ((s >> 1) & 1)) + 25 * (i2 + ((s >> 2) & 1));
        atomicAdd(&acc[((base + widx) << cshift) + c], w0 * w1 * w2 * xv);
    }
}

// ---------------- WMMA fragment loaders (LDS -> VGPR, per CDNA5 ISA layouts) ----------------

// A (16x32 f16): lanes 0-15 row M=lane, K {0..7,16..23}; lanes 16-31 row M=lane-16, K {8..15,24..31}
__device__ __forceinline__ v16h load_a_frag(const _Float16* a) {
    int lane = threadIdx.x & 31;
    int g = lane >> 4, m = lane & 15;
    union { v16h v; unsigned int u[8]; } r;
    const unsigned int* row = (const unsigned int*)(a + m * LDA);
#pragma unroll
    for (int i = 0; i < 8; ++i) {
        int k0 = ((i & 4) << 2) + (g << 3) + ((i & 3) << 1);  // {0..7,16..23} + 8g
        r.u[i] = row[k0 >> 1];
    }
    return r.v;
}

// B (32x16 f16, stored transposed in LDS as [n][k]): lanes 0-15 col N=lane K {0..7,16..23}; lanes 16-31 K {8..15,24..31}
__device__ __forceinline__ v16h load_b_frag(const _Float16* bT) {
    int lane = threadIdx.x & 31;
    int g = lane >> 4, n = lane & 15;
    union { v16h v; unsigned int u[8]; } r;
    const unsigned int* col = (const unsigned int*)(bT + n * LDA);
#pragma unroll
    for (int i = 0; i < 8; ++i) {
        int k0 = ((i & 4) << 2) + (g << 3) + ((i & 3) << 1);
        r.u[i] = col[k0 >> 1];
    }
    return r.v;
}

// pack two f32 -> one dword of two f16 (v_cvt_pk_rtz_f16_f32)
__device__ __forceinline__ unsigned int pk2(float a, float b) {
    union { v2fp16 h; unsigned int u; } r;
    r.h = __builtin_amdgcn_cvt_pkrtz(a, b);
    return r.u;
}

// ---------------- tile staging ----------------

// A tile: 64 rows x 32 K from src (row stride K f32) -> aS (f16, LDA stride)
__device__ __forceinline__ void load_tile_a(const float* __restrict__ src, int rowbase,
                                            int K, int kb, _Float16* aS, int tid, bool vec) {
    if (vec) {
        // 512 float4 / 256 threads = 2 each; K%4==0 and kb+32<=K guaranteed by caller
#pragma unroll
        for (int it = 0; it < 2; ++it) {
            int e = tid + it * 256;          // float4 index within tile
            int row = e >> 3;                // 8 float4 per 32-wide row
            int kk = (e & 7) << 2;
            const float4 v = *(const float4*)(src + (long long)(rowbase + row) * K + kb + kk);
            unsigned int* dst = (unsigned int*)(aS + row * LDA + kk);
            dst[0] = pk2(v.x, v.y);
            dst[1] = pk2(v.z, v.w);
        }
    } else {
        for (int i = tid; i < 64 * 32; i += 256) {
            int row = i >> 5, kk = i & 31;
            int k = kb + kk;
            float val = (k < K) ? src[(long long)(rowbase + row) * K + k] : 0.0f;
            aS[row * LDA + kk] = (_Float16)val;
        }
    }
}

// B tile: rows k of Wsrc (row stride out_c f32) -> bS transposed [o][kk]
__device__ __forceinline__ void load_tile_b(const float* __restrict__ Wsrc, int K, int kb,
                                            int out_c, int oshift, _Float16* bS, int tid) {
    int total4 = (out_c * 32) >> 2;          // 256 or 512 float4
    for (int e = tid; e < total4; e += 256) {
        int idx = e << 2;
        int o = idx & (out_c - 1);
        int kk = idx >> oshift;
        int k = kb + kk;
        float4 v = {0.0f, 0.0f, 0.0f, 0.0f};
        if (k < K) v = *(const float4*)(Wsrc + (long long)k * out_c + o);
        bS[(o + 0) * LDA + kk] = (_Float16)v.x;
        bS[(o + 1) * LDA + kk] = (_Float16)v.y;
        bS[(o + 2) * LDA + kk] = (_Float16)v.z;
        bS[(o + 3) * LDA + kk] = (_Float16)v.w;
    }
}

__device__ __forceinline__ void wmma_step(const _Float16* aS, const _Float16* bS,
                                          int rt, int cg, int NT, v8f& c0, v8f& c1) {
    v16h af = load_a_frag(&aS[rt * 16 * LDA]);
    v16h bf0 = load_b_frag(&bS[cg * 16 * LDA]);
    c0 = __builtin_amdgcn_wmma_f32_16x16x32_f16(false, af, false, bf0, (short)0, c0, false, false);
    if (NT == 4) {
        v16h bf1 = load_b_frag(&bS[(cg + 2) * 16 * LDA]);
        c1 = __builtin_amdgcn_wmma_f32_16x16x32_f16(false, af, false, bf1, (short)0, c1, false, false);
    }
}

// ---------------- fused spline-conv GEMM ----------------
// out[n,o] = elu( (acc[n,:] @ Wf) / max(deg,1) + x[n,:] @ R + bias )
// BM=64 rows/block, 8 waves: wave w -> row tile (w&3), col tiles (w>>2) and (w>>2)+2 (if out_c==64)

__global__ __launch_bounds__(256) void spline_gemm(
        const float* __restrict__ acc, const float* __restrict__ Wf,
        const float* __restrict__ x,   const float* __restrict__ R,
        const float* __restrict__ bias, const float* __restrict__ deg,
        float* __restrict__ out, int n, int in_c, int out_c) {
    __shared__ _Float16 aS[64 * LDA];
    __shared__ _Float16 bS[64 * LDA];

    int tid = threadIdx.x;
    int lane = tid & 31;
    int wave = tid >> 5;
    int rt = wave & 3;                        // row tile 0..3
    int cg = wave >> 2;                       // col tile group 0..1
    int NT = out_c >> 4;                      // 2 or 4 col tiles
    int oshift = (out_c == 64) ? 6 : 5;
    int rowbase = blockIdx.x * 64;

    v8f c0 = {};
    v8f c1 = {};

    // ---- pass 1: acc @ Wf, K = 125*in_c ----
    int K = 125 * in_c;
    bool vecA = ((K & 3) == 0);
    for (int kb = 0; kb < K; kb += 32) {
        load_tile_a(acc, rowbase, K, kb, aS, tid, vecA && (kb + 32 <= K));
        load_tile_b(Wf, K, kb, out_c, oshift, bS, tid);
        __syncthreads();
        wmma_step(aS, bS, rt, cg, NT, c0, c1);
        __syncthreads();
    }

    // ---- scale by 1/deg (rows of C: M = rt*16 + r + 8*(lane>=16)) ----
    {
        int g = lane >> 4;
#pragma unroll
        for (int r = 0; r < 8; ++r) {
            float d = deg[rowbase + rt * 16 + g * 8 + r];
            float s = __builtin_amdgcn_rcpf(fmaxf(d, 1.0f));
            c0[r] *= s;
            c1[r] *= s;
        }
    }

    // ---- pass 2: + x @ R, K2 = in_c ----
    bool vecX = ((in_c & 3) == 0);
    for (int kb = 0; kb < in_c; kb += 32) {
        load_tile_a(x, rowbase, in_c, kb, aS, tid, vecX && (kb + 32 <= in_c));
        load_tile_b(R, in_c, kb, out_c, oshift, bS, tid);
        __syncthreads();
        wmma_step(aS, bS, rt, cg, NT, c0, c1);
        __syncthreads();
    }

    // ---- epilogue: bias + ELU + store (col of C = lane&15, fixed per lane) ----
    {
        int g = lane >> 4, nc = lane & 15;
        int col0 = cg * 16 + nc;
        float b0v = bias[col0];
#pragma unroll
        for (int r = 0; r < 8; ++r) {
            int row = rowbase + rt * 16 + g * 8 + r;
            float v = c0[r] + b0v;
            v = (v > 0.0f) ? v : expm1f(v);
            out[(long long)row * out_c + col0] = v;
        }
        if (NT == 4) {
            int col1 = (cg + 2) * 16 + nc;
            float b1v = bias[col1];
#pragma unroll
            for (int r = 0; r < 8; ++r) {
                int row = rowbase + rt * 16 + g * 8 + r;
                float v = c1[r] + b1v;
                v = (v > 0.0f) ? v : expm1f(v);
                out[(long long)row * out_c + col1] = v;
            }
        }
    }
}

// ---------------- segment max pooling ----------------

__device__ __forceinline__ void atomicMaxF(float* addr, float val) {
    unsigned int* ua = (unsigned int*)addr;
    unsigned int old = *ua;
    while (__uint_as_float(old) < val) {
        unsigned int assumed = old;
        old = atomicCAS(ua, assumed, __float_as_uint(val));
        if (old == assumed) break;
    }
}

__global__ void pool_max(const float* __restrict__ h, const int* __restrict__ cluster,
                         float* __restrict__ outp, int n, int C) {
    int t = blockIdx.x * blockDim.x + threadIdx.x;
    if (t >= n * C) return;
    int node = t / C, c = t - node * C;
    atomicMaxF(&outp[(long long)cluster[node] * C + c], h[t]);
}

// ---------------- head: mean -> fc1 -> fc2 -> log_softmax ----------------

__global__ void head_kernel(const float* __restrict__ h,
                            const float* __restrict__ fc1w, const float* __restrict__ fc1b,
                            const float* __restrict__ fc2w, const float* __restrict__ fc2b,
                            float* __restrict__ outp) {
    __shared__ float g[64], g1[64], g2[10];
    int t = threadIdx.x;
    float s = 0.0f;
    for (int r = 0; r < 64; ++r) s += h[r * 64 + t];
    g[t] = s * (1.0f / 64.0f);
    __syncthreads();
    float a = fc1b[t];
    for (int c = 0; c < 64; ++c) a += g[c] * fc1w[c * 64 + t];
    g1[t] = a;
    __syncthreads();
    if (t < 10) {
        float b = fc2b[t];
        for (int c = 0; c < 64; ++c) b += g1[c] * fc2w[c * 10 + t];
        g2[t] = b;
    }
    __syncthreads();
    if (t < 10) {
        float m = -1e30f;
        for (int i = 0; i < 10; ++i) m = fmaxf(m, g2[i]);
        float se = 0.0f;
        for (int i = 0; i < 10; ++i) se += expf(g2[i] - m);
        outp[t] = g2[t] - m - logf(se);
    }
}

// ---------------- launcher ----------------

static inline int cdivi(long long a, int b) { return (int)((a + b - 1) / b); }
static inline int ilog2i(int v) { int s = 0; while ((1 << s) < v) ++s; return s; }

extern "C" void kernel_launch(void* const* d_in, const int* in_sizes, int n_in,
                              void* d_out, int out_size, void* d_ws, size_t ws_size,
                              hipStream_t stream) {
    const float* x0 = (const float*)d_in[0];
    const int*   src[4]     = {(const int*)d_in[1],  (const int*)d_in[5],  (const int*)d_in[9],  (const int*)d_in[13]};
    const int*   dst[4]     = {(const int*)d_in[2],  (const int*)d_in[6],  (const int*)d_in[10], (const int*)d_in[14]};
    const float* pseudo[4]  = {(const float*)d_in[3],(const float*)d_in[7],(const float*)d_in[11],(const float*)d_in[15]};
    const int*   cluster[4] = {(const int*)d_in[4],  (const int*)d_in[8],  (const int*)d_in[12], (const int*)d_in[16]};
    const float *W[8], *R[8], *B[8];
    for (int i = 0; i < 8; ++i) {
        W[i] = (const float*)d_in[17 + 3 * i];
        R[i] = (const float*)d_in[18 + 3 * i];
        B[i] = (const float*)d_in[19 + 3 * i];
    }
    const float* fc1w = (const float*)d_in[41];
    const float* fc1b = (const float*)d_in[42];
    const float* fc2w = (const float*)d_in[43];
    const float* fc2b = (const float*)d_in[44];

    char* ws = (char*)d_ws;
    float* acc = (float*)ws;                                   // up to 16384*125*32 f32 = 262 MB
    size_t off = (size_t)16384 * 125 * 32 * sizeof(float);
    float* hA  = (float*)(ws + off); off += (size_t)16384 * 64 * sizeof(float);
    float* hB  = (float*)(ws + off); off += (size_t)16384 * 64 * sizeof(float);
    float* deg = (float*)(ws + off); off += (size_t)16384 * sizeof(float);

    auto conv = [&](const float* xin, int lvl, int li, float* outp, int n, int in_c, int out_c) {
        int E = n * 8;
        long long accN = (long long)n * 125 * in_c;
        fill4_kernel<<<cdivi(accN >> 2, 256), 256, 0, stream>>>((float4*)acc, 0.0f, accN >> 2);
        fill4_kernel<<<cdivi(n >> 2, 256), 256, 0, stream>>>((float4*)deg, 0.0f, n >> 2);
        int cshift = ilog2i(in_c);
        long long total = (long long)E * in_c;
        spline_scatter<<<cdivi(total, 256), 256, 0, stream>>>(xin, src[lvl], dst[lvl], pseudo[lvl],
                                                              acc, deg, E, in_c, cshift);
        spline_gemm<<<n / 64, 256, 0, stream>>>(acc, W[li], xin, R[li], B[li], deg, outp,
                                                n, in_c, out_c);
    };
    auto pool = [&](const float* hin, int lvl, float* outp, int n, int m, int C) {
        long long mc = (long long)m * C;
        fill4_kernel<<<cdivi(mc >> 2, 256), 256, 0, stream>>>((float4*)outp, -1e30f, mc >> 2);
        pool_max<<<cdivi((long long)n * C, 256), 256, 0, stream>>>(hin, cluster[lvl], outp, n, C);
    };

    // level 0 (n=16384)
    conv(x0, 0, 0, hA, 16384, 1, 32);    // W1
    conv(hA, 0, 1, hB, 16384, 32, 32);   // W12
    pool(hB, 0, hA, 16384, 4096, 32);
    // level 1 (n=4096)
    conv(hA, 1, 2, hB, 4096, 32, 64);    // W2
    conv(hB, 1, 3, hA, 4096, 64, 64);    // W3
    pool(hA, 1, hB, 4096, 1024, 64);
    // level 2 (n=1024)
    conv(hB, 2, 4, hA, 1024, 64, 64);    // W4
    conv(hA, 2, 5, hB, 1024, 64, 64);    // W5
    pool(hB, 2, hA, 1024, 256, 64);
    // level 3 (n=256)
    conv(hA, 3, 6, hB, 256, 64, 64);     // W6
    conv(hB, 3, 7, hA, 256, 64, 64);     // W7
    pool(hA, 3, hB, 256, 64, 64);
    // head
    head_kernel<<<1, 64, 0, stream>>>(hB, fc1w, fc1b, fc2w, fc2b, (float*)d_out);
}
